// MultiHeadAttention_39316130627950
// MI455X (gfx1250) — matrix operand order repaired and checked by reading, not compiled, not run on hardware
//
#include <hip/hip_runtime.h>
#include <stdint.h>

// ---------------------------------------------------------------------------
// MultiHeadAttention forward for MI455X (gfx1250, wave32, WMMA bf16).
// B=4, S=2048, D_MODEL=512, H=8, D_K=64.
// Pipeline: mask bit-pack -> 3x bf16-WMMA GEMM (register-prefetch pipelined,
// 1/sqrt(dk) folded into Q) -> flash attention (double-buffered TDM K tiles +
// async-LDS V tiles, packed-bit mask, WMMA row-sums, online softmax)
// -> bf16-WMMA GEMM (output proj, f32 out).
// ---------------------------------------------------------------------------

typedef __attribute__((ext_vector_type(16))) __bf16 v16bf;
typedef __attribute__((ext_vector_type(8)))  __bf16 v8bf;
typedef __attribute__((ext_vector_type(8)))  float  v8f;
typedef __attribute__((ext_vector_type(4)))  unsigned int v4u;
typedef __attribute__((ext_vector_type(8)))  int v8i;
typedef __attribute__((ext_vector_type(4)))  int v4i;

#define NHEAD  8
#define DK     64
#define BATCH  4
#define SEQ    2048
#define DMODEL 512
#define MTOT   (BATCH * SEQ)

__device__ __forceinline__ v8f wmma_bf16(v16bf a, v16bf b, v8f c) {
  return __builtin_amdgcn_wmma_f32_16x16x32_bf16(false, a, false, b, (short)0, c,
                                                 false, false);
}

__device__ __forceinline__ v16bf cat8(v8bf lo, v8bf hi) {
  return __builtin_shufflevector(lo, hi, 0, 1, 2, 3, 4, 5, 6, 7,
                                 8, 9, 10, 11, 12, 13, 14, 15);
}

// A-fragment (16x32 bf16): lane<16 -> row=lane, K 0..15;
// lane>=16 -> row=lane-16, K 16..31.
__device__ __forceinline__ v16bf load_a_frag(const __bf16* base, int pitch, int lane) {
  const int half = lane >> 4, r = lane & 15;
  const __bf16* row = base + r * pitch;
  v8bf lo = *(const v8bf*)(row + half * 16);
  v8bf hi = *(const v8bf*)(row + half * 16 + 8);
  return cat8(lo, hi);
}

// B-fragment (32x16 bf16): lane holds column n=lane&15, K 0..15 / 16..31.
__device__ __forceinline__ v16bf load_b_frag(const __bf16* base, int pitch, int lane) {
  const int half = lane >> 4, n = lane & 15;
  const __bf16* p = base + n * pitch + half * 16;
  v8bf lo = *(const v8bf*)(p);
  v8bf hi = *(const v8bf*)(p + 8);
  return cat8(lo, hi);
}

// ---------------------------------------------------------------------------
// Mask bit-pack: mask[B][Sq][Sk] (u8 bool) -> Mt[B][Sk][Sq/8], bit r of byte
// (b, sk, sq0/8) = mask[b][sq0+r][sk].  One 128x128 tile per block via LDS.
// ---------------------------------------------------------------------------
__global__ __launch_bounds__(256) void mask_pack_kernel(
    const unsigned char* __restrict__ m, unsigned char* __restrict__ mt) {
  __shared__ unsigned char Ml[128][144];
  const int tid = threadIdx.x;
  const int b = blockIdx.z;
  const int sq0 = blockIdx.x * 128;
  const int sk0 = blockIdx.y * 128;

  {  // coalesced 64B-per-thread tile load
    const int row = tid >> 1, c0 = (tid & 1) * 64;
    const unsigned char* src =
        m + ((size_t)(b * SEQ + sq0 + row)) * SEQ + sk0 + c0;
#pragma unroll
    for (int i = 0; i < 4; ++i)
      *(uint4*)(&Ml[row][c0 + i * 16]) = *(const uint4*)(src + i * 16);
  }
  __syncthreads();

  const int skl = tid >> 1, sqb0 = (tid & 1) * 8;
  unsigned long long out = 0;
#pragma unroll
  for (int j = 0; j < 8; ++j) {
    unsigned byte = 0;
#pragma unroll
    for (int r = 0; r < 8; ++r)
      byte |= (unsigned)(Ml[(sqb0 + j) * 8 + r][skl] != 0) << r;
    out |= (unsigned long long)byte << (8 * j);
  }
  *(unsigned long long*)(mt + ((size_t)(b * SEQ + sk0 + skl)) * (SEQ / 8) +
                         (sq0 >> 3) + sqb0) = out;
}

// ---------------------------------------------------------------------------
// GEMM: out = (A(MxK) @ W(NxK)^T + bias) * oscale.  Tile 128x64, K-step 32.
// 256 threads = 8 waves (4x2), each wave a 32x32 block (2x2 WMMA tiles).
// Software pipeline: global loads for step k0+32 overlap the WMMAs of k0.
// ---------------------------------------------------------------------------
template <bool A_BF16, bool OUT_QKV>
__global__ __launch_bounds__(256) void gemm_wmma_kernel(
    const void* __restrict__ Ap, const float* __restrict__ W,
    const float* __restrict__ bias, void* __restrict__ Op, float oscale) {
  constexpr int K = DMODEL;
  constexpr int N = DMODEL;
  __shared__ __bf16 Asm[128][40];
  __shared__ __bf16 Wsm[64][40];

  const int tid = threadIdx.x;
  const int lane = tid & 31;
  const int wave = tid >> 5;
  const int m0 = blockIdx.x * 128;
  const int n0 = blockIdx.y * 64;
  const int wm = (wave & 3) * 32;
  const int wn = (wave >> 2) * 32;

  v8f acc[2][2] = {};

  const int arow = tid >> 1, acol = (tid & 1) * 16;
  const int wrow = tid >> 2, wcol = (tid & 3) * 8;

  // pipeline registers
  float fa[16];
  v8bf pa0, pa1;
  float wf[8];

  auto gload = [&](int k0) {
    if (A_BF16) {
      const __bf16* src = (const __bf16*)Ap + (size_t)(m0 + arow) * K + k0 + acol;
      pa0 = *(const v8bf*)(src);
      pa1 = *(const v8bf*)(src + 8);
    } else {
      const float* src = (const float*)Ap + (size_t)(m0 + arow) * K + k0 + acol;
#pragma unroll
      for (int i = 0; i < 16; ++i) fa[i] = src[i];
    }
    const float* wsrc = W + (size_t)(n0 + wrow) * K + k0 + wcol;
#pragma unroll
    for (int i = 0; i < 8; ++i) wf[i] = wsrc[i];
  };
  auto lstore = [&]() {
    if (A_BF16) {
      *(v8bf*)(&Asm[arow][acol]) = pa0;
      *(v8bf*)(&Asm[arow][acol + 8]) = pa1;
    } else {
      v8bf y0, y1;
#pragma unroll
      for (int i = 0; i < 8; ++i) {
        y0[i] = (__bf16)fa[i];
        y1[i] = (__bf16)fa[8 + i];
      }
      *(v8bf*)(&Asm[arow][acol]) = y0;
      *(v8bf*)(&Asm[arow][acol + 8]) = y1;
    }
    v8bf w;
#pragma unroll
    for (int i = 0; i < 8; ++i) w[i] = (__bf16)wf[i];
    *(v8bf*)(&Wsm[wrow][wcol]) = w;
  };

  gload(0);
  for (int k0 = 0; k0 < K; k0 += 32) {
    lstore();
    __syncthreads();
    if (k0 + 32 < K) gload(k0 + 32);  // overlaps WMMAs below

    v16bf afr[2], bfr[2];
#pragma unroll
    for (int im = 0; im < 2; ++im)
      afr[im] = load_a_frag(&Asm[wm + im * 16][0], 40, lane);
#pragma unroll
    for (int jn = 0; jn < 2; ++jn)
      bfr[jn] = load_b_frag(&Wsm[wn + jn * 16][0], 40, lane);
#pragma unroll
    for (int im = 0; im < 2; ++im)
#pragma unroll
      for (int jn = 0; jn < 2; ++jn)
        acc[im][jn] = wmma_bf16(afr[im], bfr[jn], acc[im][jn]);
    __syncthreads();
  }

  const int half = lane >> 4, cn = lane & 15;
#pragma unroll
  for (int im = 0; im < 2; ++im) {
#pragma unroll
    for (int jn = 0; jn < 2; ++jn) {
      const int gn = n0 + wn + jn * 16 + cn;
      const float bv = bias[gn];
#pragma unroll
      for (int r = 0; r < 8; ++r) {
        const int gm = m0 + wm + im * 16 + half * 8 + r;
        const float v = (acc[im][jn][r] + bv) * oscale;
        if (OUT_QKV) {
          __bf16* Oq = (__bf16*)Op;
          const int b = gm >> 11, s = gm & (SEQ - 1);
          const int h = gn >> 6, d = gn & 63;
          Oq[((((size_t)b * NHEAD + h) * SEQ + s) << 6) + d] = (__bf16)v;
        } else {
          float* Of = (float*)Op;
          Of[(size_t)gm * N + gn] = v;
        }
      }
    }
  }
}

// ---------------------------------------------------------------------------
// Address-space helpers for the async-LDS builtin (AS1 int4* / AS3 int4*),
// built with integer casts (low 32 bits of a generic LDS pointer ARE the LDS
// address per the aperture rules).
// ---------------------------------------------------------------------------
typedef int gccv4i __attribute__((__vector_size__(16)));
typedef __attribute__((address_space(1))) gccv4i gbuf128;
typedef __attribute__((address_space(3))) gccv4i lbuf128;

__device__ __forceinline__ gbuf128* as_global128(const void* p) {
  return (gbuf128*)(unsigned long long)(uintptr_t)p;
}
__device__ __forceinline__ lbuf128* as_lds128(const void* p) {
  return (lbuf128*)(unsigned)(uintptr_t)p;
}

// ---------------------------------------------------------------------------
// Tensor Data Mover: DMA rows x cols (bf16) 2D tile into LDS with 16B row pad
// (LDS pitch 72 elems).  D# fields per ISA 8.3/8.4.
// ---------------------------------------------------------------------------
#if __has_builtin(__builtin_amdgcn_tensor_load_to_lds) && \
    __has_builtin(__builtin_amdgcn_s_wait_tensorcnt)
#define USE_TDM_PATH 1
__device__ __forceinline__ void tdm_load_tile_2d(unsigned lds_off,
                                                 const __bf16* gsrc,
                                                 unsigned rows, unsigned cols,
                                                 unsigned pitch_elems) {
  const unsigned long long ga = (unsigned long long)(uintptr_t)gsrc;
  v4u g0;
  g0[0] = 1u;                                   // count=1 user descriptor
  g0[1] = lds_off;                              // lds_addr
  g0[2] = (unsigned)(ga & 0xffffffffu);         // global_addr[31:0]
  g0[3] = (unsigned)((ga >> 32) & 0x01ffffffu)  // global_addr[56:32]
          | (2u << 30);                         // type=2 ("image")
  v8i g1;
  const unsigned pad_interval = 4u;  // code 4 -> 32 DWORDs (=64 bf16) per row
  const unsigned pad_amount = 3u;    // code 3 -> 4 DWORDs (=8 bf16) pad
  g1[0] = (int)((1u << 16)           // data_size=1 -> 2 bytes
                | (1u << 20)         // pad_enable
                | (pad_interval << 22) | (pad_amount << 25));
  g1[1] = (int)((cols & 0xffffu) << 16);               // tensor_dim0[15:0]
  g1[2] = (int)(((cols >> 16) & 0xffffu)               // tensor_dim0[31:16]
                | ((unsigned)(SEQ & 0xffff) << 16));   // tensor_dim1[15:0]
  g1[3] = (int)(((unsigned)SEQ >> 16)                  // tensor_dim1[31:16]
                | ((cols & 0xffffu) << 16));           // tile_dim0
  g1[4] = (int)(rows & 0xffffu);                       // tile_dim1
  g1[5] = (int)pitch_elems;                            // tensor_dim0_stride lo
  g1[6] = 0;
  g1[7] = 0;
  v4i g2 = {0, 0, 0, 0}, g3 = {0, 0, 0, 0};
#if defined(__clang_major__) && __clang_major__ >= 23
  v8i g4 = {0, 0, 0, 0, 0, 0, 0, 0};
  __builtin_amdgcn_tensor_load_to_lds(g0, g1, g2, g3, g4, 0);
#else
  __builtin_amdgcn_tensor_load_to_lds(g0, g1, g2, g3, 0);
#endif
}
#endif

#if __has_builtin(__builtin_amdgcn_global_load_async_to_lds_b128) && \
    __has_builtin(__builtin_amdgcn_s_wait_asynccnt)
#define USE_ASYNC_LDS 1
#endif

// ---------------------------------------------------------------------------
// Flash attention: WG = one (b,h); 8 waves = 8 q-blocks of 16 rows.
// Double-buffered KV tiles: tile i+1's TDM/async DMA is issued while tile i
// is computed; waits use slack 1 so only the newest transfer is outstanding.
// ---------------------------------------------------------------------------
__global__ __launch_bounds__(256) void flash_attn_kernel(
    const __bf16* __restrict__ Q, const __bf16* __restrict__ Kb,
    const __bf16* __restrict__ Vb, const unsigned char* __restrict__ mt,
    __bf16* __restrict__ X) {
  __shared__ __bf16 Kt[2][32][72];    // K tiles [k'][d] (TDM pad pitch)
  __shared__ __bf16 Vt[2][64][40];    // V^T     [d][k]
  __shared__ __bf16 Pl[8][16][40];    // per-wave P transpose scratch
#ifdef USE_ASYNC_LDS
  __shared__ __bf16 Vraw[2][32][72];  // async-DMA staging for V tiles
#endif

  const int tid = threadIdx.x, lane = tid & 31, wave = tid >> 5;
  const int bh = blockIdx.x >> 4;
  const int qchunk = blockIdx.x & 15;
  const int b = bh >> 3;
  const int h = bh & 7;
  const int q0 = (qchunk * 8 + wave) * 16;

  const __bf16* Qp = Q + ((size_t)bh * SEQ + q0) * DK;
  const __bf16* Kp = Kb + (size_t)bh * SEQ * DK;
  const __bf16* Vp = Vb + (size_t)bh * SEQ * DK;

  const int half = lane >> 4, ln = lane & 15;

  v16bf aq[2];
#pragma unroll
  for (int t = 0; t < 2; ++t) {
    const __bf16* row = Qp + (size_t)ln * DK + t * 32;
    v8bf lo = *(const v8bf*)(row + half * 16);
    v8bf hi = *(const v8bf*)(row + half * 16 + 8);
    aq[t] = cat8(lo, hi);
  }

  v16bf ones;
#pragma unroll
  for (int i = 0; i < 16; ++i) ones[i] = (__bf16)1.0f;

  v8f acc[4] = {};
  float mrow[8], lrow[8];
#pragma unroll
  for (int r = 0; r < 8; ++r) { mrow[r] = -3.0e30f; lrow[r] = 0.0f; }

  const int krow = tid >> 3, kcol = (tid & 7) * 8;  // K coop load (fallback)
  const int vk = tid & 31, vd0 = (tid >> 5) * 8;    // V coop load/transpose

  // packed mask: byte (b, sk, (q0+half*8)/8); bit r = row q0+half*8+r
  const unsigned char* mp0 =
      mt + ((size_t)b * SEQ + ln) * (SEQ / 8) + ((q0 + half * 8) >> 3);
  const unsigned char* mp1 = mp0 + 16 * (SEQ / 8);

  // ---- prologue: stage tile 0 into buffer 0 ---------------------------
#ifdef USE_TDM_PATH
  if (tid == 0) tdm_load_tile_2d((unsigned)(uintptr_t)&Kt[0][0][0], Kp, 32u, 64u, 64u);
#else
  *(v8bf*)(&Kt[0][krow][kcol]) = *(const v8bf*)(Kp + (size_t)krow * DK + kcol);
#endif
#ifdef USE_ASYNC_LDS
  __builtin_amdgcn_global_load_async_to_lds_b128(
      as_global128(Vp + (size_t)vk * DK + vd0), as_lds128(&Vraw[0][vk][vd0]),
      0, 0);
#endif

  for (int kv0 = 0; kv0 < SEQ; kv0 += 32) {
    const int p = (kv0 >> 5) & 1, qn = p ^ 1;
    __syncthreads();  // buffer qn free: everyone done computing tile kv0-32

    if (kv0 + 32 < SEQ) {
      // issue tile i+1 into buffer qn, then wait with slack 1 for tile i
#ifdef USE_TDM_PATH
      if (tid == 0)
        tdm_load_tile_2d((unsigned)(uintptr_t)&Kt[qn][0][0],
                         Kp + (size_t)(kv0 + 32) * DK, 32u, 64u, 64u);
#else
      *(v8bf*)(&Kt[qn][krow][kcol]) =
          *(const v8bf*)(Kp + (size_t)(kv0 + 32 + krow) * DK + kcol);
#endif
#ifdef USE_ASYNC_LDS
      __builtin_amdgcn_global_load_async_to_lds_b128(
          as_global128(Vp + (size_t)(kv0 + 32 + vk) * DK + vd0),
          as_lds128(&Vraw[qn][vk][vd0]), 0, 0);
#endif
      __builtin_prefetch(Kp + (size_t)(kv0 + 32) * DK, 0, 0);
#ifdef USE_TDM_PATH
      __builtin_amdgcn_s_wait_tensorcnt(1);
#endif
#ifdef USE_ASYNC_LDS
      __builtin_amdgcn_s_wait_asynccnt(1);
#endif
    } else {
#ifdef USE_TDM_PATH
      __builtin_amdgcn_s_wait_tensorcnt(0);
#endif
#ifdef USE_ASYNC_LDS
      __builtin_amdgcn_s_wait_asynccnt(0);
#endif
    }

    // ---- transpose V tile p into Vt[p] -------------------------------
    {
#ifdef USE_ASYNC_LDS
      v8bf vx = *(const v8bf*)(&Vraw[p][vk][vd0]);
#else
      v8bf vx = *(const v8bf*)(Vp + (size_t)(kv0 + vk) * DK + vd0);
#endif
#pragma unroll
      for (int j = 0; j < 8; ++j) Vt[p][vd0 + j][vk] = vx[j];
    }
    const unsigned mb0 = *mp0, mb1 = *mp1;
    mp0 += 32 * (SEQ / 8);
    mp1 += 32 * (SEQ / 8);
    __syncthreads();  // Kt[p] (wave0's tensor wait) + Vt[p] visible to all

    // ---- scores S = Q K^T (scale pre-folded into Q) ------------------
    v8f sc[2];
#pragma unroll
    for (int t2 = 0; t2 < 2; ++t2) {
      v8f s = {};
#pragma unroll
      for (int db = 0; db < 2; ++db) {
        const __bf16* pk = &Kt[p][t2 * 16 + ln][db * 32 + half * 16];
        s = wmma_bf16(aq[db], cat8(*(const v8bf*)pk, *(const v8bf*)(pk + 8)), s);
      }
      sc[t2] = s;
    }

    // ---- masked fill from packed bits --------------------------------
#pragma unroll
    for (int r = 0; r < 8; ++r) {
      if ((mb0 >> r) & 1) sc[0][r] = -10000.0f;
      if ((mb1 >> r) & 1) sc[1][r] = -10000.0f;
    }

    // ---- online softmax: max via xor-shuffles, sum via WMMA ----------
    float alpha[8];
#pragma unroll
    for (int r = 0; r < 8; ++r) {
      float mx = fmaxf(sc[0][r], sc[1][r]);
#pragma unroll
      for (int off = 8; off >= 1; off >>= 1)
        mx = fmaxf(mx, __shfl_xor(mx, off, 32));
      const float mnew = fmaxf(mrow[r], mx);
      alpha[r] = __expf(mrow[r] - mnew);
      mrow[r] = mnew;
      sc[0][r] = __expf(sc[0][r] - mnew);
      sc[1][r] = __expf(sc[1][r] - mnew);
#pragma unroll
      for (int db = 0; db < 4; ++db) acc[db][r] *= alpha[r];
    }

    // ---- transpose P through wave-private LDS into an A-fragment -----
#pragma unroll
    for (int t2 = 0; t2 < 2; ++t2)
#pragma unroll
      for (int r = 0; r < 8; ++r)
        Pl[wave][half * 8 + r][t2 * 16 + ln] = (__bf16)sc[t2][r];
    asm volatile("s_wait_dscnt 0" ::: "memory");
    const v16bf aP = load_a_frag(&Pl[wave][0][0], 40, lane);

    // ---- row sums: P x ones (one WMMA instead of 32 shuffles) --------
    {
      v8f z = {};
      const v8f rs = wmma_bf16(aP, ones, z);
#pragma unroll
      for (int r = 0; r < 8; ++r) lrow[r] = lrow[r] * alpha[r] + rs[r];
    }

    // ---- O += P V ----------------------------------------------------
#pragma unroll
    for (int db = 0; db < 4; ++db) {
      const __bf16* pv = &Vt[p][db * 16 + ln][half * 16];
      acc[db] =
          wmma_bf16(aP, cat8(*(const v8bf*)pv, *(const v8bf*)(pv + 8)), acc[db]);
    }
  }

  // ---- epilogue: O /= l, write bf16 X[b][s][h*64+d] -------------------
#pragma unroll
  for (int db = 0; db < 4; ++db) {
    const int d = db * 16 + ln;
#pragma unroll
    for (int r = 0; r < 8; ++r) {
      const int s = q0 + half * 8 + r;
      X[((size_t)b * SEQ + s) * DMODEL + h * DK + d] =
          (__bf16)(acc[db][r] / lrow[r]);
    }
  }
}

// ---------------------------------------------------------------------------
extern "C" void kernel_launch(void* const* d_in, const int* in_sizes, int n_in,
                              void* d_out, int out_size, void* d_ws,
                              size_t ws_size, hipStream_t stream) {
  (void)in_sizes; (void)n_in; (void)out_size; (void)ws_size;
  const float* query = (const float*)d_in[0];
  const float* key = (const float*)d_in[1];
  const float* value = (const float*)d_in[2];
  const unsigned char* mask = (const unsigned char*)d_in[3];
  const float* Wq = (const float*)d_in[4];
  const float* bq = (const float*)d_in[5];
  const float* Wk = (const float*)d_in[6];
  const float* bk = (const float*)d_in[7];
  const float* Wv = (const float*)d_in[8];
  const float* bv = (const float*)d_in[9];
  const float* Wo = (const float*)d_in[10];
  const float* bo = (const float*)d_in[11];

  char* ws = (char*)d_ws;
  const size_t seg = (size_t)BATCH * NHEAD * SEQ * DK * sizeof(__bf16);  // 8 MB
  __bf16* Qb = (__bf16*)(ws);
  __bf16* Kb = (__bf16*)(ws + seg);
  __bf16* Vb = (__bf16*)(ws + 2 * seg);
  __bf16* Xb = (__bf16*)(ws + 3 * seg);
  unsigned char* Mt = (unsigned char*)(ws + 4 * seg);  // 2 MB packed mask

  mask_pack_kernel<<<dim3(SEQ / 128, SEQ / 128, BATCH), 256, 0, stream>>>(mask,
                                                                          Mt);

  dim3 gg(MTOT / 128, DMODEL / 64);
  gemm_wmma_kernel<false, true><<<gg, 256, 0, stream>>>(query, Wq, bq, Qb,
                                                        0.125f);
  gemm_wmma_kernel<false, true><<<gg, 256, 0, stream>>>(key, Wk, bk, Kb, 1.0f);
  gemm_wmma_kernel<false, true><<<gg, 256, 0, stream>>>(value, Wv, bv, Vb,
                                                        1.0f);

  flash_attn_kernel<<<BATCH * NHEAD * (SEQ / 128), 256, 0, stream>>>(
      Qb, Kb, Vb, Mt, Xb);

  gemm_wmma_kernel<true, false><<<gg, 256, 0, stream>>>(Xb, Wo, bo,
                                                        (float*)d_out, 1.0f);
}
